// TransformerEncoder_27814208209359
// MI455X (gfx1250) — matrix operand order repaired and checked by reading, not compiled, not run on hardware
//
#include <hip/hip_runtime.h>
#include <hip/hip_bf16.h>

typedef __attribute__((ext_vector_type(16))) _Float16 v16h;
typedef __attribute__((ext_vector_type(8)))  _Float16 v8h;
typedef __attribute__((ext_vector_type(8)))  float    v8f;

union V16U { v16h v; v8h h[2]; };

#define NLAYER 12
#define NHEAD  12
#define DMODEL 768
#define BATCH  32
#define NTOK   576
#define DHEAD  64
#define QKVW   (3 * DMODEL)   // 2304

// ---------------------------------------------------------------------------
// CDNA5 async global->LDS copy (bypasses VGPRs, tracked by ASYNCcnt).
// Low 32 bits of a generic shared pointer are the LDS byte offset.
// ---------------------------------------------------------------------------
__device__ __forceinline__ void async_g2l_b128(void* lds_dst, const void* gsrc) {
  unsigned lds_off = (unsigned)(unsigned long long)lds_dst;
  unsigned long long ga = (unsigned long long)gsrc;
  asm volatile("global_load_async_to_lds_b128 %0, %1, off"
               :: "v"(lds_off), "v"(ga)
               : "memory");
}
__device__ __forceinline__ void wait_async() {
  asm volatile("s_wait_asynccnt 0x0" ::: "memory");
}

// ---------------------------------------------------------------------------
// f32 -> f16 elementwise convert (weights, once per call)
// ---------------------------------------------------------------------------
__global__ __launch_bounds__(256) void cvt_f16_kernel(const float* __restrict__ in,
                                                      _Float16* __restrict__ out, int n) {
  int i = blockIdx.x * 256 + threadIdx.x;
  if (i < n) out[i] = (_Float16)in[i];
}

// ---------------------------------------------------------------------------
// LayerNorm over D=768, one row per 256-thread block; emits f16 activations
// ---------------------------------------------------------------------------
__global__ __launch_bounds__(256) void ln_kernel(const float* __restrict__ z,
                                                 const float* __restrict__ g,
                                                 const float* __restrict__ b,
                                                 _Float16* __restrict__ out) {
  const int row = blockIdx.x;                       // b*N + n
  const float* x = z + (size_t)row * DMODEL;
  const int t = threadIdx.x;
  float v0 = x[t], v1 = x[t + 256], v2 = x[t + 512];

  float s = v0 + v1 + v2;
#pragma unroll
  for (int off = 16; off >= 1; off >>= 1) s += __shfl_xor(s, off, 32);
  __shared__ float red1[8], red2[8];
  const int wid = t >> 5, lid = t & 31;
  if (lid == 0) red1[wid] = s;
  __syncthreads();
  float tot = red1[0];
#pragma unroll
  for (int w = 1; w < 8; w++) tot += red1[w];
  const float mean = tot * (1.0f / DMODEL);

  float d0 = v0 - mean, d1 = v1 - mean, d2 = v2 - mean;
  float sq = d0 * d0 + d1 * d1 + d2 * d2;
#pragma unroll
  for (int off = 16; off >= 1; off >>= 1) sq += __shfl_xor(sq, off, 32);
  if (lid == 0) red2[wid] = sq;
  __syncthreads();
  float vtot = red2[0];
#pragma unroll
  for (int w = 1; w < 8; w++) vtot += red2[w];
  const float rstd = rsqrtf(vtot * (1.0f / DMODEL) + 1e-5f);

  _Float16* o = out + (size_t)row * DMODEL;
  o[t]       = (_Float16)(d0 * rstd * g[t]       + b[t]);
  o[t + 256] = (_Float16)(d1 * rstd * g[t + 256] + b[t + 256]);
  o[t + 512] = (_Float16)(d2 * rstd * g[t + 512] + b[t + 512]);
}

// ---------------------------------------------------------------------------
// WMMA GEMM: C[M x Ncols] = A[M x 768] * B[768 x Ncols], f16 in, f32 acc.
// Block tile 64(M) x 128(N), 8 waves (2x4), wave tile 32x32, K step 32.
// Staging uses async global->LDS (no VGPR round trip).
// MODE 0: store f16 result (QKV).  MODE 1: z += C + bias (MLP + residual).
// ---------------------------------------------------------------------------
template <int MODE>
__global__ __launch_bounds__(256) void gemm_wmma_kernel(
    const _Float16* __restrict__ A, const _Float16* __restrict__ Bw,
    _Float16* __restrict__ Cf16, float* __restrict__ Zres,
    const float* __restrict__ bias, int Ncols) {
  __shared__ _Float16 As[64][40];        // 32 used, padded (80B rows, 16B aligned)
  __shared__ _Float16 Bs[32][136];       // 128 used, padded (272B rows)

  const int tid  = threadIdx.x;
  const int lane = tid & 31;
  const int wave = tid >> 5;
  const int mw = wave >> 2;              // 0..1
  const int nw = wave & 3;               // 0..3
  const int hi  = lane >> 4;             // lane 16..31 -> 1
  const int l15 = lane & 15;
  const int tileM = blockIdx.y * 64;
  const int tileN = blockIdx.x * 128;

  v8f acc[2][2] = {};

  for (int k0 = 0; k0 < DMODEL; k0 += 32) {
    {   // stage A tile: 64x32 halfs, one async b128 per thread
      int r = tid >> 2, c = (tid & 3) * 8;
      async_g2l_b128(&As[r][c], A + (size_t)(tileM + r) * DMODEL + k0 + c);
    }
    {   // stage B tile: 32x128 halfs, two async b128 per thread
      int r = tid >> 3, c = (tid & 7) * 16;
      const _Float16* src = Bw + (size_t)(k0 + r) * Ncols + tileN + c;
      async_g2l_b128(&Bs[r][c],     src);
      async_g2l_b128(&Bs[r][c + 8], src + 8);
    }
    wait_async();
    __syncthreads();

    v16h a[2], bf[2];
#pragma unroll
    for (int mi = 0; mi < 2; mi++) {     // A frag: row = lane&15, kk per ISA layout
      const _Float16* arow = &As[mw * 32 + mi * 16 + l15][hi * 8];
      v16h tv;
#pragma unroll
      for (int p = 0; p < 4; p++) {
        tv[2 * p]         = arow[2 * p];
        tv[2 * p + 1]     = arow[2 * p + 1];
        tv[8 + 2 * p]     = arow[16 + 2 * p];
        tv[8 + 2 * p + 1] = arow[16 + 2 * p + 1];
      }
      a[mi] = tv;
    }
#pragma unroll
    for (int ni = 0; ni < 2; ni++) {     // B frag: col = lane&15, K = (i&7) + (i>>3)*16 + hi*8
      const int col = nw * 32 + ni * 16 + l15;
      v16h tv;
#pragma unroll
      for (int i = 0; i < 16; i++) tv[i] = Bs[(i & 7) + (i >> 3) * 16 + hi * 8][col];
      bf[ni] = tv;
    }
#pragma unroll
    for (int mi = 0; mi < 2; mi++)
#pragma unroll
      for (int ni = 0; ni < 2; ni++)
        acc[mi][ni] = __builtin_amdgcn_wmma_f32_16x16x32_f16(
            false, a[mi], false, bf[ni], (short)0, acc[mi][ni], false, false);
    __syncthreads();
  }

#pragma unroll
  for (int mi = 0; mi < 2; mi++)
#pragma unroll
    for (int ni = 0; ni < 2; ni++) {
      const int grow0 = tileM + mw * 32 + mi * 16 + hi * 8;
      const int gcol  = tileN + nw * 32 + ni * 16 + l15;
#pragma unroll
      for (int r = 0; r < 8; r++) {
        const size_t idx = (size_t)(grow0 + r) * Ncols + gcol;
        if (MODE == 0) Cf16[idx] = (_Float16)acc[mi][ni][r];
        else           Zres[idx] += acc[mi][ni][r] + bias[gcol];
      }
    }
}

// ---------------------------------------------------------------------------
// Flash attention, one wave per (b, h, 16-query tile). qkv layout:
// [B*N][h*192 + s*64 + d] (matches reference reshape (H,3,Dh)).
// Online softmax; residual-accumulates straight into z.
// ---------------------------------------------------------------------------
__global__ __launch_bounds__(32) void attn_kernel(const _Float16* __restrict__ qkv,
                                                  float* __restrict__ z) {
  __shared__ _Float16 pbuf[16][40];      // P tile 16x32 (C-layout -> A-layout repack)
  __shared__ _Float16 vbuf[32][72];      // V chunk 32x64 (144B rows, 16B aligned)

  const int lane = threadIdx.x & 31;
  const int hi = lane >> 4, l15 = lane & 15;
  int bid = blockIdx.x;                  // B*H*36
  const int qt = bid % 36; bid /= 36;
  const int h  = bid % NHEAD;
  const int bb = bid / NHEAD;
  const int qbase = qt * 16;
  const _Float16* base = qkv + (size_t)bb * NTOK * QKVW + h * (3 * DHEAD);

  // Q fragments for dh chunks 0 / 1 (A layout, 16x32 each)
  v16h qa[2];
  {
    const _Float16* qrow = base + (size_t)(qbase + l15) * QKVW;
#pragma unroll
    for (int c = 0; c < 2; c++) {
      const _Float16* src = qrow + c * 32 + hi * 8;
      v16h tv;
#pragma unroll
      for (int p = 0; p < 4; p++) {
        tv[2 * p]         = src[2 * p];
        tv[2 * p + 1]     = src[2 * p + 1];
        tv[8 + 2 * p]     = src[16 + 2 * p];
        tv[8 + 2 * p + 1] = src[16 + 2 * p + 1];
      }
      qa[c] = tv;
    }
  }

  v8f o[4] = {};
  float M[8], Lsum[8];
#pragma unroll
  for (int r = 0; r < 8; r++) { M[r] = -1e30f; Lsum[r] = 0.0f; }
  const float scale = 0.125f;            // 1/sqrt(64)

  for (int j = 0; j < NTOK / 32; j++) {  // 18 chunks of 32 keys
    const int jb = j * 32;

    // V chunk 32x64 -> LDS via async DMA, one row per lane (overlaps with S/softmax)
    {
      const _Float16* vrow = base + 2 * DHEAD + (size_t)(jb + lane) * QKVW;
#pragma unroll
      for (int p = 0; p < 8; p++)
        async_g2l_b128(&vbuf[lane][p * 8], vrow + p * 8);
    }

    // S = Q*K^T for two 16-key subtiles
    v8f s[2] = {};
#pragma unroll
    for (int t2 = 0; t2 < 2; t2++) {
      const _Float16* krow = base + DHEAD + (size_t)(jb + t2 * 16 + l15) * QKVW;
#pragma unroll
      for (int c = 0; c < 2; c++) {
        const _Float16* src = krow + c * 32 + hi * 8;    // two 8-half runs at +0 and +16
        V16U kb;
        kb.h[0] = *(const v8h*)(src);
        kb.h[1] = *(const v8h*)(src + 16);
        s[t2] = __builtin_amdgcn_wmma_f32_16x16x32_f16(
            false, qa[c], false, kb.v, (short)0, s[t2], false, false);
      }
    }

    // online softmax update (rows live in 16-lane halves -> width-16 shuffles)
    float p0[8], p1[8], alpha[8];
#pragma unroll
    for (int r = 0; r < 8; r++) {
      float x0 = s[0][r] * scale, x1 = s[1][r] * scale;
      float mx = fmaxf(x0, x1);
#pragma unroll
      for (int off = 8; off >= 1; off >>= 1) mx = fmaxf(mx, __shfl_xor(mx, off, 16));
      const float nm = fmaxf(M[r], mx);
      alpha[r] = __expf(M[r] - nm);
      p0[r] = __expf(x0 - nm);
      p1[r] = __expf(x1 - nm);
      float rs = p0[r] + p1[r];
#pragma unroll
      for (int off = 8; off >= 1; off >>= 1) rs += __shfl_xor(rs, off, 16);
      Lsum[r] = Lsum[r] * alpha[r] + rs;
      M[r] = nm;
    }

    // P (C-layout) -> LDS row-major f16
#pragma unroll
    for (int r = 0; r < 8; r++) {
      pbuf[r + hi * 8][l15]      = (_Float16)p0[r];
      pbuf[r + hi * 8][16 + l15] = (_Float16)p1[r];
    }
    wait_async();
    __syncthreads();

    // P as A fragment (16x32)
    v16h pa;
    {
      const _Float16* prow = &pbuf[l15][hi * 8];
#pragma unroll
      for (int p = 0; p < 4; p++) {
        pa[2 * p]         = prow[2 * p];
        pa[2 * p + 1]     = prow[2 * p + 1];
        pa[8 + 2 * p]     = prow[16 + 2 * p];
        pa[8 + 2 * p + 1] = prow[16 + 2 * p + 1];
      }
    }
    // O = O*alpha + P*V  (4 B fragments cover dh 0..63)
#pragma unroll
    for (int nt = 0; nt < 4; nt++) {
      v16h vb;
#pragma unroll
      for (int i = 0; i < 16; i++) vb[i] = vbuf[(i & 7) + (i >> 3) * 16 + hi * 8][nt * 16 + l15];
#pragma unroll
      for (int r = 0; r < 8; r++) o[nt][r] *= alpha[r];
      o[nt] = __builtin_amdgcn_wmma_f32_16x16x32_f16(
          false, pa, false, vb, (short)0, o[nt], false, false);
    }
    __syncthreads();
  }

  // z += O / L  (each element owned by exactly one lane slot)
#pragma unroll
  for (int nt = 0; nt < 4; nt++)
#pragma unroll
    for (int r = 0; r < 8; r++) {
      const size_t idx = ((size_t)bb * NTOK + qbase + r + hi * 8) * DMODEL
                         + h * DHEAD + nt * 16 + l15;
      z[idx] += o[nt][r] / Lsum[r];
    }
}

// ---------------------------------------------------------------------------
extern "C" void kernel_launch(void* const* d_in, const int* in_sizes, int n_in,
                              void* d_out, int out_size, void* d_ws, size_t ws_size,
                              hipStream_t stream) {
  (void)in_sizes; (void)n_in; (void)out_size; (void)ws_size;
  const float* z_in  = (const float*)d_in[0];
  const float* Wqkv  = (const float*)d_in[1];
  const float* ln1_g = (const float*)d_in[2];
  const float* ln1_b = (const float*)d_in[3];
  const float* Wmlp  = (const float*)d_in[4];
  const float* bmlp  = (const float*)d_in[5];
  const float* ln2_g = (const float*)d_in[6];
  const float* ln2_b = (const float*)d_in[7];
  float* z = (float*)d_out;

  // workspace layout (f16 elements)
  const size_t nWqkv = (size_t)NLAYER * DMODEL * QKVW;     // 21.2M
  const size_t nWmlp = (size_t)NLAYER * DMODEL * DMODEL;   // 7.1M
  const size_t nXh   = (size_t)BATCH * NTOK * DMODEL;      // 14.2M
  _Float16* wqkv_h = (_Float16*)d_ws;
  _Float16* wmlp_h = wqkv_h + nWqkv;
  _Float16* xh     = wmlp_h + nWmlp;
  _Float16* qkvh   = xh + nXh;                             // B*N*2304

  (void)hipMemcpyAsync(z, z_in, (size_t)BATCH * NTOK * DMODEL * sizeof(float),
                       hipMemcpyDeviceToDevice, stream);

  cvt_f16_kernel<<<(int)((nWqkv + 255) / 256), 256, 0, stream>>>(Wqkv, wqkv_h, (int)nWqkv);
  cvt_f16_kernel<<<(int)((nWmlp + 255) / 256), 256, 0, stream>>>(Wmlp, wmlp_h, (int)nWmlp);

  const int nRows = BATCH * NTOK;                          // 18432
  for (int l = 0; l < NLAYER; l++) {
    // --- MSA: z += Attn(LN1(z)) ---
    ln_kernel<<<nRows, 256, 0, stream>>>(z, ln1_g + l * DMODEL, ln1_b + l * DMODEL, xh);
    gemm_wmma_kernel<0><<<dim3(QKVW / 128, nRows / 64), 256, 0, stream>>>(
        xh, wqkv_h + (size_t)l * DMODEL * QKVW, qkvh, nullptr, nullptr, QKVW);
    attn_kernel<<<BATCH * NHEAD * (NTOK / 16), 32, 0, stream>>>(qkvh, z);
    // --- MLP: z += LN2(z) @ Wmlp + bmlp ---
    ln_kernel<<<nRows, 256, 0, stream>>>(z, ln2_g + l * DMODEL, ln2_b + l * DMODEL, xh);
    gemm_wmma_kernel<1><<<dim3(DMODEL / 128, nRows / 64), 256, 0, stream>>>(
        xh, wmlp_h + (size_t)l * DMODEL * DMODEL, nullptr, z, bmlp + l * DMODEL, DMODEL);
  }
}
